// HybridConvLayer_88072599371912
// MI455X (gfx1250) — compile-verified, matrix operand order checked by reading, not run on hardware
//
#include <hip/hip_runtime.h>

#define DD 128          // feature dim
#define KTOT 896        // 7 * 128 concat dim
#define KC 32           // K chunk per WMMA step
#define NCHUNK (KTOT / KC)   // 28
#define BROWS 64        // output rows per block
#define LDSTR 40        // padded LDS row stride in bf16 elems (80B: 16B aligned segments, no bank conflicts)

typedef __attribute__((ext_vector_type(8)))  float  v8f;
typedef __attribute__((ext_vector_type(8)))  __bf16 v8bf;
typedef __attribute__((ext_vector_type(2)))  __bf16 v2bf;
typedef __attribute__((ext_vector_type(16))) __bf16 v16bf;

__device__ __forceinline__ __bf16 f2bf(float f) {
  union { float f; unsigned u; } a; a.f = f;
  unsigned r = a.u + 0x7FFFu + ((a.u >> 16) & 1u);   // round to nearest even
  union { unsigned short s; __bf16 b; } o; o.s = (unsigned short)(r >> 16);
  return o.b;
}

// pack 8 floats -> v8bf, using hardware V_CVT_PK_BF16_F32 when available
__device__ __forceinline__ v8bf pack8bf(const float* v) {
  v8bf pk;
#if __has_builtin(__builtin_amdgcn_cvt_pk_bf16_f32)
  union { v8bf v; v2bf p[4]; } u;
#pragma unroll
  for (int j = 0; j < 4; ++j)
    u.p[j] = __builtin_amdgcn_cvt_pk_bf16_f32(v[2 * j], v[2 * j + 1]);
  pk = u.v;
#else
#pragma unroll
  for (int j = 0; j < 8; ++j) pk[j] = f2bf(v[j]);
#endif
  return pk;
}

__device__ __forceinline__ void ld8(const float* __restrict__ p, float* o) {
  const float4* q = (const float4*)p;
  float4 u = q[0], v = q[1];
  o[0]=u.x; o[1]=u.y; o[2]=u.z; o[3]=u.w;
  o[4]=v.x; o[5]=v.y; o[6]=v.z; o[7]=v.w;
}

// ---------------- utility kernels ----------------

__global__ void k_zero(float* __restrict__ p, int n) {
  int i = blockIdx.x * blockDim.x + threadIdx.x;
  if (i < n) p[i] = 0.f;
}

__global__ void k_deg(const long long* __restrict__ ei, float* __restrict__ deg, int E) {
  int e = blockIdx.x * blockDim.x + threadIdx.x;
  if (e < E) {
    long long d = ei[(long long)E + e];
    __hip_atomic_fetch_add(&deg[d], 1.0f, __ATOMIC_RELAXED, __HIP_MEMORY_SCOPE_AGENT);
  }
}

__global__ void k_wedge(const long long* __restrict__ ei, const float* __restrict__ deg,
                        float* __restrict__ we, int E) {
  int e = blockIdx.x * blockDim.x + threadIdx.x;
  if (e < E) {
    float dg = deg[ei[(long long)E + e]];
    we[e] = dg > 0.f ? 0.5f / dg : 0.f;
  }
}

// one-time: W[896][128] fp32 -> Wt[128][896] bf16 (transposed: row = output col, contiguous in K)
__global__ void k_wbf(const float* __restrict__ W, __bf16* __restrict__ Wt, int n) {
  int i = blockIdx.x * blockDim.x + threadIdx.x;   // i indexes Wt flat
  if (i < n) {
    int k = i % KTOT;
    int c = i / KTOT;
    Wt[i] = f2bf(W[(long long)k * DD + c]);
  }
}

// out = 0.5 * h  (self-loop term, also initializes accumulator)
__global__ void k_half(const float* __restrict__ h, float* __restrict__ out, int n4) {
  int i = blockIdx.x * blockDim.x + threadIdx.x;
  if (i < n4) {
    float4 v = ((const float4*)h)[i];
    v.x *= 0.5f; v.y *= 0.5f; v.z *= 0.5f; v.w *= 0.5f;
    ((float4*)out)[i] = v;
  }
}

// one wave per edge: lane l handles features 4l..4l+3; atomics stay L2-resident (51MB << 192MB)
__global__ __launch_bounds__(256) void k_scatter(const long long* __restrict__ ei,
                                                 const float* __restrict__ we,
                                                 const float* __restrict__ h,
                                                 float* __restrict__ out, int E) {
  int gid = blockIdx.x * 256 + threadIdx.x;
  int e = gid >> 5;
  if (e >= E) return;
  int lane = gid & 31;
  long long s = ei[e];
  long long d = ei[(long long)E + e];
  float w = we[e];
  float4 hv = ((const float4*)(h + s * DD))[lane];
  float* op = out + d * DD + (long long)lane * 4;
  __hip_atomic_fetch_add(op + 0, w * hv.x, __ATOMIC_RELAXED, __HIP_MEMORY_SCOPE_AGENT);
  __hip_atomic_fetch_add(op + 1, w * hv.y, __ATOMIC_RELAXED, __HIP_MEMORY_SCOPE_AGENT);
  __hip_atomic_fetch_add(op + 2, w * hv.z, __ATOMIC_RELAXED, __HIP_MEMORY_SCOPE_AGENT);
  __hip_atomic_fetch_add(op + 3, w * hv.w, __ATOMIC_RELAXED, __HIP_MEMORY_SCOPE_AGENT);
}

// one wave per node: 6 inverse L2 norms (a1, a2, a4, x-a1, a1-a2, a2-a4)
__global__ __launch_bounds__(256) void k_norms(const float* __restrict__ x,
                                               const float* __restrict__ a1,
                                               const float* __restrict__ a2,
                                               const float* __restrict__ a4,
                                               float* __restrict__ invn, int N) {
  int node = (blockIdx.x * 256 + threadIdx.x) >> 5;
  int lane = threadIdx.x & 31;
  if (node >= N) return;
  long long base = (long long)node * DD + (long long)lane * 4;
  float4 vx = *(const float4*)(x  + base);
  float4 v1 = *(const float4*)(a1 + base);
  float4 v2 = *(const float4*)(a2 + base);
  float4 v4 = *(const float4*)(a4 + base);
  float xs[4] = {vx.x, vx.y, vx.z, vx.w};
  float s1[4] = {v1.x, v1.y, v1.z, v1.w};
  float s2[4] = {v2.x, v2.y, v2.z, v2.w};
  float s4[4] = {v4.x, v4.y, v4.z, v4.w};
  float s[6] = {0.f, 0.f, 0.f, 0.f, 0.f, 0.f};
#pragma unroll
  for (int j = 0; j < 4; ++j) {
    float d1 = xs[j] - s1[j], d2 = s1[j] - s2[j], d3 = s2[j] - s4[j];
    s[0] += s1[j] * s1[j];
    s[1] += s2[j] * s2[j];
    s[2] += s4[j] * s4[j];
    s[3] += d1 * d1;
    s[4] += d2 * d2;
    s[5] += d3 * d3;
  }
#pragma unroll
  for (int off = 16; off > 0; off >>= 1) {
#pragma unroll
    for (int c = 0; c < 6; ++c) s[c] += __shfl_xor(s[c], off, 32);
  }
  if (lane == 0) {
#pragma unroll
    for (int c = 0; c < 6; ++c)
      invn[(long long)node * 6 + c] = 1.f / fmaxf(sqrtf(s[c]), 1e-12f);
  }
}

// ---------------- fused normalize+concat GEMM with bf16 WMMA ----------------
// C[N,128] = leakyrelu(hcat[N,896] @ W[896,128] + b)
// A tiles synthesized on the fly from {x, a1, a2, a4, invnorm} -> bf16 LDS (double-buffered).
// B fragments loaded directly from pre-transposed bf16 Wt[128][896] in global (L2-resident).

// stage one 64x32 bf16 A tile; c is a compile-time constant after unrolling
__device__ __forceinline__ void stage_a(
    int c, int kk,
    const float* __restrict__ x,  const float* __restrict__ a1,
    const float* __restrict__ a2, const float* __restrict__ a4,
    const float* nv6, bool inb, long long rowbase, int r, int kl,
    __bf16 (*dst)[LDSTR]) {
  float v[8];
  if (inb) {
    const float* pA;
    const float* pB = nullptr;
    float s = 1.f;
    switch (c) {
      case 0:  pA = x;                          break;
      case 1:  pA = a1;           s = nv6[0];   break;
      case 2:  pA = a2;           s = nv6[1];   break;
      case 3:  pA = a4;           s = nv6[2];   break;
      case 4:  pA = x;  pB = a1;  s = nv6[3];   break;
      case 5:  pA = a1; pB = a2;  s = nv6[4];   break;
      default: pA = a2; pB = a4;  s = nv6[5];   break;
    }
    const long long base = rowbase + kk + kl;
    float ta[8];
    ld8(pA + base, ta);
    if (pB) {
      float tb[8];
      ld8(pB + base, tb);
#pragma unroll
      for (int j = 0; j < 8; ++j) v[j] = (ta[j] - tb[j]) * s;
    } else {
#pragma unroll
      for (int j = 0; j < 8; ++j) v[j] = ta[j] * s;
    }
  } else {
#pragma unroll
    for (int j = 0; j < 8; ++j) v[j] = 0.f;
  }
  *(v8bf*)&dst[r][kl] = pack8bf(v);    // one 16B ds_store per thread per chunk
}

__global__ __launch_bounds__(256) void k_gemm(const float* __restrict__ x,
                                              const float* __restrict__ a1,
                                              const float* __restrict__ a2,
                                              const float* __restrict__ a4,
                                              const float* __restrict__ invn,
                                              const __bf16* __restrict__ Wt,
                                              const float* __restrict__ bias,
                                              float* __restrict__ out, int N) {
  __shared__ __align__(16) __bf16 lds_a[2][BROWS][LDSTR];  // double-buffered A tile (2 x 5KB)

  const int tid  = threadIdx.x;
  const int wave = tid >> 5;          // 0..7 -> owns N-tile cols 16w..16w+15
  const int lane = tid & 31;
  const int half = lane >> 4;         // WMMA lane-half
  const int nl   = lane & 15;         // M (A) / N (B,C) index within tile
  const int koff = half * 8;          // K phase per ISA bf16 layout
  const int row0 = blockIdx.x * BROWS;

  // A-staging role for this thread
  const int r  = tid >> 2;            // 0..63
  const int kl = (tid & 3) * 8;       // 0,8,16,24
  const int node = row0 + r;
  const bool inb = node < N;
  const long long rowbase = (long long)node * DD;

  // hoist the 6 inverse norms for this thread's staged node
  float nv6[6] = {1.f, 1.f, 1.f, 1.f, 1.f, 1.f};
  if (inb) {
    const float* p = invn + (long long)node * 6;
#pragma unroll
    for (int c = 0; c < 6; ++c) nv6[c] = p[c];
  }

  // per-lane base of this lane's B row in Wt (contiguous in K)
  const __bf16* wtrow = Wt + (long long)(wave * 16 + nl) * KTOT;

  v8f acc[4] = {v8f{}, v8f{}, v8f{}, v8f{}};

  stage_a(0, 0, x, a1, a2, a4, nv6, inb, rowbase, r, kl, lds_a[0]);
  __syncthreads();

#pragma unroll
  for (int i = 0; i < NCHUNK; ++i) {
    // prefetch next chunk into the other buffer (c, kk fold to constants after unroll)
    if (i + 1 < NCHUNK) {
      stage_a((i + 1) >> 2, ((i + 1) & 3) * KC,
              x, a1, a2, a4, nv6, inb, rowbase, r, kl, lds_a[(i + 1) & 1]);
    }

    // B fragment straight from global (two global_load_b128, L2/WGP$-resident)
    union { v16bf v; v8bf h[2]; } ub;
    const int k0 = i * KC;
    ub.h[0] = *(const v8bf*)(wtrow + k0 + koff);
    ub.h[1] = *(const v8bf*)(wtrow + k0 + koff + 16);

    // A fragments from LDS per ISA bf16 layout; 4 M-tiles
    const __bf16 (*buf)[LDSTR] = lds_a[i & 1];
#pragma unroll
    for (int t = 0; t < 4; ++t) {
      union { v16bf v; v8bf h[2]; } ua;
      const int ar = t * 16 + nl;
      ua.h[0] = *(const v8bf*)&buf[ar][koff];
      ua.h[1] = *(const v8bf*)&buf[ar][koff + 16];
      acc[t] = __builtin_amdgcn_wmma_f32_16x16x32_bf16(
          false, ua.v, false, ub.v, (short)0, acc[t], false, false);
    }
    if (i + 1 < NCHUNK) __syncthreads();   // one barrier per chunk (ping-pong)
  }

  // ---- epilogue: bias + leaky-relu, per documented C/D layout ----
  const int col = wave * 16 + nl;
  const float bc = bias[col];
#pragma unroll
  for (int t = 0; t < 4; ++t) {
#pragma unroll
    for (int vr = 0; vr < 8; ++vr) {
      const int row = row0 + t * 16 + half * 8 + vr;
      if (row < N) {
        float v = acc[t][vr] + bc;
        out[(long long)row * DD + col] = v > 0.f ? v : 0.01f * v;
      }
    }
  }
}

// ---------------- launch ----------------

extern "C" void kernel_launch(void* const* d_in, const int* in_sizes, int n_in,
                              void* d_out, int out_size, void* d_ws, size_t ws_size,
                              hipStream_t stream) {
  const float*     x  = (const float*)d_in[0];
  const long long* ei = (const long long*)d_in[1];   // int64 edge_index [2,E]
  const float*     Wc = (const float*)d_in[2];
  const float*     bc = (const float*)d_in[3];
  float* out = (float*)d_out;

  const int N = in_sizes[0] / DD;
  const int E = in_sizes[1] / 2;
  const size_t ND = (size_t)N * DD;

  float* deg  = (float*)d_ws;
  float* we   = deg + N;
  float* a1   = we + E;
  float* a2   = a1 + ND;
  float* tmp3 = a2 + ND;
  float* a4   = tmp3 + ND;
  float* invn = a4 + ND;
  __bf16* Wt  = (__bf16*)(invn + (size_t)N * 6);

  dim3 blk(256);
  const int gE  = (E + 255) / 256;
  const int n4  = (int)(ND / 4);
  const int g4  = (n4 + 255) / 256;
  const int gS  = (int)(((long long)E * 32 + 255) / 256);  // one wave per edge

  k_zero<<<(N + 255) / 256, blk, 0, stream>>>(deg, N);
  k_deg<<<gE, blk, 0, stream>>>(ei, deg, E);
  k_wedge<<<gE, blk, 0, stream>>>(ei, deg, we, E);
  k_wbf<<<(KTOT * DD + 255) / 256, blk, 0, stream>>>(Wc, Wt, KTOT * DD);

  // a1 = P x ; a2 = P a1 ; tmp3 = P a2 ; a4 = P tmp3
  k_half<<<g4, blk, 0, stream>>>(x, a1, n4);
  k_scatter<<<gS, blk, 0, stream>>>(ei, we, x, a1, E);
  k_half<<<g4, blk, 0, stream>>>(a1, a2, n4);
  k_scatter<<<gS, blk, 0, stream>>>(ei, we, a1, a2, E);
  k_half<<<g4, blk, 0, stream>>>(a2, tmp3, n4);
  k_scatter<<<gS, blk, 0, stream>>>(ei, we, a2, tmp3, E);
  k_half<<<g4, blk, 0, stream>>>(tmp3, a4, n4);
  k_scatter<<<gS, blk, 0, stream>>>(ei, we, tmp3, a4, E);

  k_norms<<<((size_t)N * 32 + 255) / 256, blk, 0, stream>>>(x, a1, a2, a4, invn, N);
  k_gemm<<<(N + BROWS - 1) / BROWS, blk, 0, stream>>>(x, a1, a2, a4, invn, Wt, bc, out, N);
}
